// MLP_17798344475235
// MI455X (gfx1250) — compile-verified, matrix operand order checked by reading, not run on hardware
//
#include <hip/hip_runtime.h>
#include <stdint.h>

// Problem geometry (fixed by the reference).
#define M_TOK 4096   // 2 * 2048 tokens
#define D_IN  4096
#define H_FF  11008
#define GROUPQ 128   // quant group along in-features

// Tile geometry.
#define BM  128
#define BN  64
#define BK  64       // K-chunk staged per iteration (half a quant group -> z/s constant)
#define LDK 72       // padded LDS row stride in halves (144 B, 16B aligned)

typedef __attribute__((ext_vector_type(4)))  float    v4f;
typedef __attribute__((ext_vector_type(4)))  _Float16 v4h;
typedef __attribute__((ext_vector_type(8)))  _Float16 v8h;
typedef __attribute__((ext_vector_type(16))) _Float16 v16h;
typedef __attribute__((ext_vector_type(8)))  float    v8f;

// Load one 16x32 f16 WMMA fragment's per-lane data from an LDS row:
// lane<16 holds K runs {0..7, 16..23}, lane>=16 holds {8..15, 24..31}.
__device__ __forceinline__ v16h frag_ld(const _Float16* p) {
  v8h lo = *(const v8h*)(p);        // ds_load_b128
  v8h hi = *(const v8h*)(p + 16);   // ds_load_b128 (+32 bytes)
  return __builtin_shufflevector(lo, hi, 0, 1, 2, 3, 4, 5, 6, 7,
                                 8, 9, 10, 11, 12, 13, 14, 15);
}

// Async-copy one BM x BK f16 tile from global into LDS (row stride LDK halves).
// 4 global_load_async_to_lds_b128 per wave -> ASYNCcnt += 4.
__device__ __forceinline__ void async_a_stage(const _Float16* __restrict__ src,
                                              size_t src_stride, _Float16* dst,
                                              int tid) {
  for (int c = tid; c < BM * (BK / 8); c += 256) {
    int row = c >> 3;
    int off = (c & 7) << 3;
    unsigned lds = (unsigned)(uintptr_t)(dst + row * LDK + off);
    const void* g = (const void*)(src + (size_t)row * src_stride + off);
    asm volatile("global_load_async_to_lds_b128 %0, %1, off"
                 :: "v"(lds), "v"(g) : "memory");
  }
}

__global__ __launch_bounds__(256)
void cvt_f32_f16(const v4f* __restrict__ x, v4h* __restrict__ y, int n4) {
  int i = blockIdx.x * blockDim.x + threadIdx.x;
  if (i < n4) y[i] = __builtin_convertvector(x[i], v4h);
}

// Fused gate+up w4a16 GEMM with SwiGLU epilogue: h = silu(x Wg) * (x Wu), f16 out.
__global__ __launch_bounds__(256)
void gateup_w4a16(const _Float16* __restrict__ xh,
                  const int* __restrict__ qw_gate, const int* __restrict__ z_gate,
                  const float* __restrict__ s_gate,
                  const int* __restrict__ qw_up,   const int* __restrict__ z_up,
                  const float* __restrict__ s_up,
                  _Float16* __restrict__ hout) {
  __shared__ _Float16 As[2][BM * LDK];  // double-buffered, filled by async engine
  __shared__ _Float16 Bg[BN * LDK];     // column-major [n][k], dequantized f16
  __shared__ _Float16 Bu[BN * LDK];

  const int tid   = threadIdx.x;
  const int lane  = tid & 31;
  const int lhalf = lane & 15;
  const int hi16  = lane >> 4;
  const int wave  = tid >> 5;
  const int wm    = wave & 3;    // 4 waves along M (32 rows each)
  const int wn    = wave >> 2;   // 2 waves along N (32 cols each)

  // M fastest in the grid: the 32 M-blocks sharing this N-tile's weights are
  // coresident, so int4 weights stream HBM->L2 ~once; xh stays L2-resident.
  const int m0 = blockIdx.x * BM;
  const int n0 = blockIdx.y * BN;
  const int ncol = tid & 63;     // each thread dequantizes one fixed column

  v8f accG[2][2] = {};
  v8f accU[2][2] = {};

  const int NCHUNK = D_IN / BK;
  async_a_stage(&xh[(size_t)m0 * D_IN], D_IN, &As[0][0], tid);  // prologue

  for (int ck = 0; ck < NCHUNK; ++ck) {
    const int cur = ck & 1, nxt = cur ^ 1;
    const int k0  = ck * BK;
    const int grp = k0 / GROUPQ;

    // Kick off the NEXT chunk's A tile while we dequantize/compute this one.
    if (ck + 1 < NCHUNK)
      async_a_stage(&xh[(size_t)m0 * D_IN + (k0 + BK)], D_IN, &As[nxt][0], tid);

    // Per-column dequant params, constant over this K-chunk (GROUP=128 >= BK).
    const float zg = (float)z_gate[(size_t)grp * H_FF + n0 + ncol];
    const float sg = s_gate[(size_t)grp * H_FF + n0 + ncol];
    const float zu = (float)z_up[(size_t)grp * H_FF + n0 + ncol];
    const float su = s_up[(size_t)grp * H_FF + n0 + ncol];

    // Dequant both weight tiles into LDS, column-major so fragment loads are
    // contiguous-in-K b128 LDS loads. Global reads are n-contiguous (coalesced).
#pragma unroll
    for (int j = 0; j < BK / 4; ++j) {
      int k  = (tid >> 6) + (j << 2);
      int qg = qw_gate[(size_t)(k0 + k) * H_FF + n0 + ncol];
      int qu = qw_up  [(size_t)(k0 + k) * H_FF + n0 + ncol];
      Bg[ncol * LDK + k] = (_Float16)(((float)qg - zg) * sg);
      Bu[ncol * LDK + k] = (_Float16)(((float)qu - zu) * su);
    }
    if (ck + 1 < NCHUNK) {
      __builtin_prefetch(&qw_gate[(size_t)(k0 + BK) * H_FF + n0 + ncol], 0, 1);
      __builtin_prefetch(&qw_up  [(size_t)(k0 + BK) * H_FF + n0 + ncol], 0, 1);
      // Only wait for the CURRENT tile's 4 async ops (in-order completion);
      // the 4 just issued for the next tile stay in flight under the WMMAs.
      asm volatile("s_wait_asynccnt 0x4" ::: "memory");
    } else {
      asm volatile("s_wait_asynccnt 0x0" ::: "memory");
    }
    __syncthreads();   // A[cur] + dequantized B visible to all waves

#pragma unroll
    for (int kk = 0; kk < BK / 32; ++kk) {
      v16h a[2], bg[2], bu[2];
#pragma unroll
      for (int i = 0; i < 2; ++i)
        a[i] = frag_ld(&As[cur][(wm * 32 + i * 16 + lhalf) * LDK + kk * 32 + hi16 * 8]);
#pragma unroll
      for (int j = 0; j < 2; ++j) {
        bg[j] = frag_ld(&Bg[(wn * 32 + j * 16 + lhalf) * LDK + kk * 32 + hi16 * 8]);
        bu[j] = frag_ld(&Bu[(wn * 32 + j * 16 + lhalf) * LDK + kk * 32 + hi16 * 8]);
      }
#pragma unroll
      for (int i = 0; i < 2; ++i)
#pragma unroll
        for (int j = 0; j < 2; ++j) {
          accG[i][j] = __builtin_amdgcn_wmma_f32_16x16x32_f16(
              false, a[i], false, bg[j], (short)0, accG[i][j], false, false);
          accU[i][j] = __builtin_amdgcn_wmma_f32_16x16x32_f16(
              false, a[i], false, bu[j], (short)0, accU[i][j], false, false);
        }
    }
    __syncthreads();   // all waves done reading As[cur]/Bg/Bu before overwrite
  }

  // SwiGLU epilogue; C layout: VGPR r -> M = r + 8*hi16, N = lane&15.
#pragma unroll
  for (int i = 0; i < 2; ++i)
#pragma unroll
    for (int j = 0; j < 2; ++j) {
      int n  = n0 + wn * 32 + j * 16 + lhalf;
      int mb = m0 + wm * 32 + i * 16 + hi16 * 8;
#pragma unroll
      for (int r = 0; r < 8; ++r) {
        float g = accG[i][j][r];
        float u = accU[i][j][r];
        float sig = 1.0f / (1.0f + __expf(-g));
        hout[(size_t)(mb + r) * H_FF + n] = (_Float16)(g * sig * u);
      }
    }
}

// out = h Wd, w4a16 GEMM, f32 output.
__global__ __launch_bounds__(256)
void down_w4a16(const _Float16* __restrict__ hin,
                const int* __restrict__ qw_down, const int* __restrict__ z_down,
                const float* __restrict__ s_down,
                float* __restrict__ out) {
  __shared__ _Float16 As[2][BM * LDK];
  __shared__ _Float16 Bd[BN * LDK];

  const int tid   = threadIdx.x;
  const int lane  = tid & 31;
  const int lhalf = lane & 15;
  const int hi16  = lane >> 4;
  const int wave  = tid >> 5;
  const int wm    = wave & 3;
  const int wn    = wave >> 2;

  const int m0 = blockIdx.x * BM;   // M fastest: weight tiles shared in L2
  const int n0 = blockIdx.y * BN;
  const int ncol = tid & 63;

  v8f acc[2][2] = {};

  const int NCHUNK = H_FF / BK;
  async_a_stage(&hin[(size_t)m0 * H_FF], H_FF, &As[0][0], tid);

  for (int ck = 0; ck < NCHUNK; ++ck) {
    const int cur = ck & 1, nxt = cur ^ 1;
    const int k0  = ck * BK;
    const int grp = k0 / GROUPQ;

    if (ck + 1 < NCHUNK)
      async_a_stage(&hin[(size_t)m0 * H_FF + (k0 + BK)], H_FF, &As[nxt][0], tid);

    const float zd = (float)z_down[(size_t)grp * D_IN + n0 + ncol];
    const float sd = s_down[(size_t)grp * D_IN + n0 + ncol];

#pragma unroll
    for (int j = 0; j < BK / 4; ++j) {
      int k = (tid >> 6) + (j << 2);
      int q = qw_down[(size_t)(k0 + k) * D_IN + n0 + ncol];
      Bd[ncol * LDK + k] = (_Float16)(((float)q - zd) * sd);
    }
    if (ck + 1 < NCHUNK) {
      __builtin_prefetch(&qw_down[(size_t)(k0 + BK) * D_IN + n0 + ncol], 0, 1);
      asm volatile("s_wait_asynccnt 0x4" ::: "memory");
    } else {
      asm volatile("s_wait_asynccnt 0x0" ::: "memory");
    }
    __syncthreads();

#pragma unroll
    for (int kk = 0; kk < BK / 32; ++kk) {
      v16h a[2], b[2];
#pragma unroll
      for (int i = 0; i < 2; ++i)
        a[i] = frag_ld(&As[cur][(wm * 32 + i * 16 + lhalf) * LDK + kk * 32 + hi16 * 8]);
#pragma unroll
      for (int j = 0; j < 2; ++j)
        b[j] = frag_ld(&Bd[(wn * 32 + j * 16 + lhalf) * LDK + kk * 32 + hi16 * 8]);
#pragma unroll
      for (int i = 0; i < 2; ++i)
#pragma unroll
        for (int j = 0; j < 2; ++j)
          acc[i][j] = __builtin_amdgcn_wmma_f32_16x16x32_f16(
              false, a[i], false, b[j], (short)0, acc[i][j], false, false);
    }
    __syncthreads();
  }

#pragma unroll
  for (int i = 0; i < 2; ++i)
#pragma unroll
    for (int j = 0; j < 2; ++j) {
      int n  = n0 + wn * 32 + j * 16 + lhalf;
      int mb = m0 + wm * 32 + i * 16 + hi16 * 8;
#pragma unroll
      for (int r = 0; r < 8; ++r)
        out[(size_t)(mb + r) * D_IN + n] = acc[i][j][r];
    }
}

extern "C" void kernel_launch(void* const* d_in, const int* in_sizes, int n_in,
                              void* d_out, int out_size, void* d_ws, size_t ws_size,
                              hipStream_t stream) {
  (void)in_sizes; (void)n_in; (void)out_size; (void)ws_size;

  const float* x       = (const float*)d_in[0];
  const int*   qw_gate = (const int*)  d_in[1];
  const int*   z_gate  = (const int*)  d_in[2];
  const float* s_gate  = (const float*)d_in[3];
  const int*   qw_up   = (const int*)  d_in[4];
  const int*   z_up    = (const int*)  d_in[5];
  const float* s_up    = (const float*)d_in[6];
  const int*   qw_down = (const int*)  d_in[7];
  const int*   z_down  = (const int*)  d_in[8];
  const float* s_down  = (const float*)d_in[9];
  float*       out     = (float*)d_out;

  // Workspace layout: x in f16 (32 MB), then h in f16 (86 MB).
  _Float16* xh   = (_Float16*)d_ws;
  _Float16* hbuf = (_Float16*)((char*)d_ws + (size_t)M_TOK * D_IN * sizeof(_Float16));

  // 1) x f32 -> f16
  {
    int n4 = (M_TOK * D_IN) / 4;
    cvt_f32_f16<<<(n4 + 255) / 256, 256, 0, stream>>>((const v4f*)x, (v4h*)xh, n4);
  }
  // 2) fused gate/up + SwiGLU (grid: M fastest for L2 weight sharing)
  {
    dim3 grid(M_TOK / BM, H_FF / BN);   // 32 x 172
    gateup_w4a16<<<grid, 256, 0, stream>>>(xh, qw_gate, z_gate, s_gate,
                                           qw_up, z_up, s_up, hbuf);
  }
  // 3) down projection
  {
    dim3 grid(M_TOK / BM, D_IN / BN);   // 32 x 64
    down_w4a16<<<grid, 256, 0, stream>>>(hbuf, qw_down, z_down, s_down, out);
  }
}
